// LinearAttention_62758062129477
// MI455X (gfx1250) — matrix operand order repaired and checked by reading, not compiled, not run on hardware
//
#include <hip/hip_runtime.h>

typedef __bf16 bf16;
typedef __attribute__((ext_vector_type(2)))  __bf16 v2bf;
typedef __attribute__((ext_vector_type(8)))  __bf16 v8bf;
typedef __attribute__((ext_vector_type(16))) __bf16 v16bf;
typedef __attribute__((ext_vector_type(8)))  float  v8f;

constexpr int N_SEQ  = 2048;
constexpr int DHEAD  = 128;
constexpr int QTILE  = 16;   // q rows per wave
constexpr int KCHUNK = 32;   // keys per staged chunk (= WMMA K for 2nd matmul)
constexpr int WAVES  = 4;    // waves per block
constexpr int QBLOCK = QTILE * WAVES;  // 64 q rows per block

static __device__ __forceinline__ v16bf cat8(v8bf lo, v8bf hi) {
    return __builtin_shufflevector(lo, hi, 0,1,2,3,4,5,6,7,8,9,10,11,12,13,14,15);
}

__global__ __launch_bounds__(WAVES * 32)
void alibi_attn_kernel(const float* __restrict__ q,
                       const float* __restrict__ k,
                       const float* __restrict__ v,
                       float* __restrict__ out)
{
    // LDS staging: K chunk row-major bf16, V chunk transposed bf16, per-wave S scratch bf16
    __shared__ __align__(32) bf16 lds_k [KCHUNK][DHEAD];         // 8 KB
    __shared__ __align__(32) bf16 lds_vt[DHEAD][KCHUNK];         // 8 KB
    __shared__ __align__(32) bf16 lds_s [WAVES][QTILE][KCHUNK];  // 4 KB

    const int tid    = threadIdx.x;
    const int lane   = tid & 31;
    const int wave   = tid >> 5;
    const int laneHi = lane >> 4;      // 0 | 1 (lane half)
    const int laneLo = lane & 15;

    const int head  = blockIdx.y;
    const int bh    = blockIdx.z * 8 + head;
    const int qBase = blockIdx.x * QBLOCK;
    const int qRow0 = qBase + wave * QTILE;

    const float slope = exp2f(-(float)(head + 1));  // 2^-(h+1)

    const float* qp = q   + (size_t)bh * N_SEQ * DHEAD;
    const float* kp = k   + (size_t)bh * N_SEQ * DHEAD;
    const float* vp = v   + (size_t)bh * N_SEQ * DHEAD;
    float*       op = out + (size_t)bh * N_SEQ * DHEAD;

    // Row factors: em[vv] = exp(-slope * m), m = vv + 8*laneHi (C-layout row of this lane)
    float em[8];
    #pragma unroll
    for (int vv = 0; vv < 8; ++vv)
        em[vv] = __expf(-slope * (float)(vv + 8 * laneHi));

    // ---- Load Q A-fragments (4 slices of K=32 along d), ISA A-layout ----
    // lane<16: row M=laneLo, K in {0..7, 16..23}; lane>=16: K in {8..15, 24..31}
    v16bf aq[4];
    {
        const int   khalf = laneHi ? 8 : 0;
        const float* qrow = qp + (size_t)(qRow0 + laneLo) * DHEAD;
        #pragma unroll
        for (int s = 0; s < 4; ++s) {
            v16bf a;
            #pragma unroll
            for (int t = 0; t < 8; ++t) {
                a[t]     = (bf16)qrow[s * 32 + khalf + t];
                a[8 + t] = (bf16)qrow[s * 32 + 16 + khalf + t];
            }
            aq[s] = a;
        }
    }

    // O accumulators: 8 C-tiles of 16x16 f32 covering d=128
    v8f acc[8];
    #pragma unroll
    for (int t = 0; t < 8; ++t) acc[t] = (v8f){};

    const int nChunks = (qBase + QBLOCK) / KCHUNK;  // causal bound for whole block
    for (int c = 0; c < nChunks; ++c) {
        const int kBase = c * KCHUNK;

        // ---- cooperative staging ----
        // Each thread: key pair (kp2, kp2+1) x 16 d-values.
        //   K rows stay contiguous (vectorizable stores);
        //   V transpose packs (key, key+1) pairs into single b32 stores.
        {
            const int kp2  = (tid >> 3) * 2;     // even key in 0..30
            const int dseg = (tid & 7) * 16;     // d segment
            const float* krowA = kp + (size_t)(kBase + kp2) * DHEAD + dseg;
            const float* krowB = krowA + DHEAD;
            const float* vrowA = vp + (size_t)(kBase + kp2) * DHEAD + dseg;
            const float* vrowB = vrowA + DHEAD;
            #pragma unroll
            for (int i2 = 0; i2 < 16; ++i2) {
                lds_k[kp2][dseg + i2]     = (bf16)krowA[i2];
                lds_k[kp2 + 1][dseg + i2] = (bf16)krowB[i2];
                v2bf pr;
                pr[0] = (bf16)vrowA[i2];
                pr[1] = (bf16)vrowB[i2];
                *(v2bf*)&lds_vt[dseg + i2][kp2] = pr;
            }
        }
        __syncthreads();

        // wave-uniform causal skip: does this chunk touch rows <= qRow0+15 ?
        if (kBase <= qRow0 + QTILE - 1) {
            // ---- S = Q * K^T : two 16x16 tiles (keys kBase..kBase+31) ----
            #pragma unroll
            for (int tile = 0; tile < 2; ++tile) {
                v8f s = (v8f){};
                const int keyRow = tile * 16 + laneLo;     // B column n = key index
                const int dHalf  = laneHi ? 8 : 0;         // B K-half per lane half
                #pragma unroll
                for (int sl = 0; sl < 4; ++sl) {
                    const bf16* bp = &lds_k[keyRow][sl * 32 + dHalf];
                    v16bf bfrag = cat8(*(const v8bf*)bp, *(const v8bf*)(bp + 16));
                    s = __builtin_amdgcn_wmma_f32_16x16x32_bf16(
                            false, aq[sl], false, bfrag, (short)0, s, false, false);
                }
                // decay mask in f32, convert once, park in per-wave bf16 LDS scratch
                const int   j  = kBase + tile * 16 + laneLo;
                const float fb = __expf(-slope * (float)(qRow0 - j)); // bounded: <= e^{7.5}
                #pragma unroll
                for (int vv = 0; vv < 8; ++vv) {
                    const int i = qRow0 + vv + 8 * laneHi;
                    const float val = (j <= i) ? s[vv] * em[vv] * fb : 0.0f;
                    lds_s[wave][vv + 8 * laneHi][tile * 16 + laneLo] = (bf16)val;
                }
            }

            // ---- reload S as A-fragment (16 x 32 bf16): two aligned 16B loads, no cvt ----
            v16bf a2;
            {
                const int   k0 = laneHi ? 8 : 0;
                const bf16* sp = &lds_s[wave][laneLo][0];
                a2 = cat8(*(const v8bf*)(sp + k0), *(const v8bf*)(sp + 16 + k0));
            }

            // ---- O += S * V : 8 n-tiles over d ----
            const int kHalf = laneHi ? 8 : 0;
            #pragma unroll
            for (int t = 0; t < 8; ++t) {
                const int  dcol = t * 16 + laneLo;
                const bf16* vb  = &lds_vt[dcol][kHalf];
                v16bf bfrag = cat8(*(const v8bf*)vb, *(const v8bf*)(vb + 16));
                acc[t] = __builtin_amdgcn_wmma_f32_16x16x32_bf16(
                            false, a2, false, bfrag, (short)0, acc[t], false, false);
            }
        }
        __syncthreads();
    }

    // ---- write O (C-layout: lane holds (m = vv + 8*laneHi, n = laneLo) per tile) ----
    #pragma unroll
    for (int t = 0; t < 8; ++t) {
        #pragma unroll
        for (int vv = 0; vv < 8; ++vv) {
            const int m    = vv + 8 * laneHi;
            const int dcol = t * 16 + laneLo;
            op[(size_t)(qRow0 + m) * DHEAD + dcol] = acc[t][vv];
        }
    }
}

extern "C" void kernel_launch(void* const* d_in, const int* in_sizes, int n_in,
                              void* d_out, int out_size, void* d_ws, size_t ws_size,
                              hipStream_t stream) {
    (void)in_sizes; (void)n_in; (void)out_size; (void)d_ws; (void)ws_size;
    const float* q = (const float*)d_in[0];
    const float* k = (const float*)d_in[1];
    const float* v = (const float*)d_in[2];
    float* out = (float*)d_out;

    dim3 grid(N_SEQ / QBLOCK, 8, 2);   // (q-blocks, heads, batch)
    alibi_attn_kernel<<<grid, WAVES * 32, 0, stream>>>(q, k, v, out);
}
